// TselfATTENTION_24412594110530
// MI455X (gfx1250) — matrix operand order repaired and checked by reading, not compiled, not run on hardware
//
#include <hip/hip_runtime.h>

typedef __bf16 bf16;
typedef __attribute__((ext_vector_type(16))) __bf16 v16bf;
typedef __attribute__((ext_vector_type(8)))  __bf16 v8bf;
typedef __attribute__((ext_vector_type(8)))  float  v8f;

static constexpr int Bsz = 16;
static constexpr int C   = 256;
static constexpr int NT  = 1024;   // H*W tokens

// ---------------------------------------------------------------------------
// WMMA fragment loaders (wave32, v_wmma_f32_16x16x32_bf16).
// A (16x32, MxK), row-major, per ISA 7.12.2 "16-bit A-Matrix 16x32":
//   lane L (M = L%16): halves 0..7 -> K = base..base+7, halves 8..15 -> K = base+16..base+23
//   base = 0 for lanes 0..15, 8 for lanes 16..31.
// B (32x16, KxN) loaded from a transposed (NxK row-major) buffer, same K order
// as A: lane L (N = L%16): halves 0..7 -> K = base+h, 8..15 -> K = base+16+(h-8).
// C/D (16x16 f32): vgpr r, lane L -> M = r + (L<16?0:8), N = L%16.
// ---------------------------------------------------------------------------
__device__ __forceinline__ v16bf load_frag_a(const bf16* __restrict__ p0, int ld) {
  int lane = threadIdx.x & 31;
  int m  = lane & 15;
  int kb = (lane & 16) ? 8 : 0;
  const bf16* p = p0 + (size_t)m * ld + kb;
  v8bf lo = *(const v8bf*)(p);
  v8bf hi = *(const v8bf*)(p + 16);
  v16bf r;
#pragma unroll
  for (int i = 0; i < 8; ++i) { r[i] = lo[i]; r[i + 8] = hi[i]; }
  return r;
}

__device__ __forceinline__ v16bf load_frag_bt(const bf16* __restrict__ p0, int ld) {
  int lane = threadIdx.x & 31;
  int n  = lane & 15;
  int kb = (lane & 16) ? 8 : 0;
  const bf16* p = p0 + (size_t)n * ld + kb;
  v8bf lo = *(const v8bf*)(p);
  v8bf hi = *(const v8bf*)(p + 16);
  v16bf r;
#pragma unroll
  for (int i = 0; i < 8; ++i) { r[i] = lo[i]; r[i + 8] = hi[i]; }
  return r;
}

// ---------------------------------------------------------------------------
// Generic batched GEMM:  out = A[M,K] * Bt[N,K]^T (+bias) with epilogue EPI
//   EPI 0: +bias, store bf16 [M,N]
//   EPI 1: +bias, store bf16 transposed [N,M]   (for V^T)
//   EPI 2: no bias, store bf16 [M,N]            (scores, P*V)
//   EPI 3: +bias, SiLU, store bf16 [M,N]        (MLP layer 1)
//   EPI 4: +bias, store f32 [M,N]               (MLP layer 2, feeds LayerNorm)
// Block: 128 threads = 4 waves (2 in M x 2 in N). Each wave computes a 32x64
// tile (2 A-frags x 4 B-frags -> 8 WMMAs per 32-wide K-step), so a block
// covers 64 rows x 128 cols.
// ---------------------------------------------------------------------------
template <int EPI>
__global__ __launch_bounds__(128)
void gemm_bt_kernel(const bf16* __restrict__ A, const bf16* __restrict__ Bt,
                    const float* __restrict__ bias, void* __restrict__ out,
                    int K, int lda, int ldb, int ldc,
                    long long sA, long long sB, long long sC) {
  const long long b = blockIdx.z;
  A  += b * sA;
  Bt += b * sB;
  const int m0 = blockIdx.y * 64;
  const int n0 = blockIdx.x * 128;
  const int w  = threadIdx.x >> 5;
  const int mw = m0 + (w >> 1) * 32;  // 32 rows per wave
  const int nw = n0 + (w & 1) * 64;   // 64 cols per wave

  v8f acc[2][4] = {};

  for (int k0 = 0; k0 < K; k0 += 32) {
    v16bf a0 = load_frag_a(A + (size_t)mw * lda + k0, lda);
    v16bf a1 = load_frag_a(A + (size_t)(mw + 16) * lda + k0, lda);
    v16bf bt0 = load_frag_bt(Bt + (size_t)(nw +  0) * ldb + k0, ldb);
    v16bf bt1 = load_frag_bt(Bt + (size_t)(nw + 16) * ldb + k0, ldb);
    v16bf bt2 = load_frag_bt(Bt + (size_t)(nw + 32) * ldb + k0, ldb);
    v16bf bt3 = load_frag_bt(Bt + (size_t)(nw + 48) * ldb + k0, ldb);
    acc[0][0] = __builtin_amdgcn_wmma_f32_16x16x32_bf16(false, a0, false, bt0, (short)0, acc[0][0], false, false);
    acc[0][1] = __builtin_amdgcn_wmma_f32_16x16x32_bf16(false, a0, false, bt1, (short)0, acc[0][1], false, false);
    acc[0][2] = __builtin_amdgcn_wmma_f32_16x16x32_bf16(false, a0, false, bt2, (short)0, acc[0][2], false, false);
    acc[0][3] = __builtin_amdgcn_wmma_f32_16x16x32_bf16(false, a0, false, bt3, (short)0, acc[0][3], false, false);
    acc[1][0] = __builtin_amdgcn_wmma_f32_16x16x32_bf16(false, a1, false, bt0, (short)0, acc[1][0], false, false);
    acc[1][1] = __builtin_amdgcn_wmma_f32_16x16x32_bf16(false, a1, false, bt1, (short)0, acc[1][1], false, false);
    acc[1][2] = __builtin_amdgcn_wmma_f32_16x16x32_bf16(false, a1, false, bt2, (short)0, acc[1][2], false, false);
    acc[1][3] = __builtin_amdgcn_wmma_f32_16x16x32_bf16(false, a1, false, bt3, (short)0, acc[1][3], false, false);
  }

  const int lane = threadIdx.x & 31;
  const int cl = lane & 15;
  const int ro = (lane & 16) ? 8 : 0;
#pragma unroll
  for (int i = 0; i < 2; ++i)
#pragma unroll
    for (int j = 0; j < 4; ++j)
#pragma unroll
      for (int r = 0; r < 8; ++r) {
        int row = mw + i * 16 + ro + r;
        int col = nw + j * 16 + cl;
        float v = acc[i][j][r];
        if (EPI != 2) v += bias[col];
        if (EPI == 3) v = v * (1.0f / (1.0f + __expf(-v)));  // SiLU
        if (EPI == 4) {
          ((float*)out)[b * sC + (size_t)row * ldc + col] = v;
        } else if (EPI == 1) {
          ((bf16*)out)[b * sC + (size_t)col * ldc + row] = (bf16)v;
        } else {
          ((bf16*)out)[b * sC + (size_t)row * ldc + col] = (bf16)v;
        }
      }
}

// ---------------------------------------------------------------------------
// x[B,C,N] f32  ->  t[B,N,C] bf16, LDS-tiled 32x32 transpose.
// ---------------------------------------------------------------------------
__global__ __launch_bounds__(256)
void xpose_kernel(const float* __restrict__ x, bf16* __restrict__ t) {
  __shared__ float tile[32][33];
  const int b  = blockIdx.z;
  const int n0 = blockIdx.x * 32;
  const int c0 = blockIdx.y * 32;
  const int tx = threadIdx.x & 31;
  const int ty = threadIdx.x >> 5;  // 0..7
  const float* xb = x + ((size_t)b * C + c0) * NT + n0;
#pragma unroll
  for (int j = 0; j < 4; ++j)
    tile[ty + j * 8][tx] = xb[(size_t)(ty + j * 8) * NT + tx];
  __syncthreads();
  bf16* tb = t + ((size_t)b * NT + n0) * C + c0;
#pragma unroll
  for (int j = 0; j < 4; ++j)
    tb[(size_t)(ty + j * 8) * C + tx] = (bf16)tile[tx][ty + j * 8];
}

// ---------------------------------------------------------------------------
// Pack 5 weights [C,C] f32 ([in,out]) -> bf16 transposed ([out,in]) so GEMM
// B-fragments are contiguous per lane.
// ---------------------------------------------------------------------------
__global__ __launch_bounds__(256)
void wpack_kernel(const float* __restrict__ w0, const float* __restrict__ w1,
                  const float* __restrict__ w2, const float* __restrict__ w3,
                  const float* __restrict__ w4, bf16* __restrict__ out) {
  const float* ws[5] = {w0, w1, w2, w3, w4};
  const int m   = blockIdx.y;
  const int idx = blockIdx.x * 256 + threadIdx.x;  // co*256 + ci
  const int co  = idx >> 8;
  const int ci  = idx & 255;
  out[(size_t)m * (C * C) + idx] = (bf16)ws[m][ci * C + co];
}

// ---------------------------------------------------------------------------
// In-place row softmax over length-1024 rows of bf16 scores.
// ---------------------------------------------------------------------------
__global__ __launch_bounds__(256)
void softmax_kernel(bf16* __restrict__ S) {
  __shared__ float red[256];
  bf16* p = S + (size_t)blockIdx.x * NT;
  const int tid = threadIdx.x;
  float v[4];
#pragma unroll
  for (int t = 0; t < 4; ++t) v[t] = (float)p[tid + 256 * t];
  float m = fmaxf(fmaxf(v[0], v[1]), fmaxf(v[2], v[3]));
  red[tid] = m;
  __syncthreads();
  for (int s = 128; s > 0; s >>= 1) {
    if (tid < s) red[tid] = fmaxf(red[tid], red[tid + s]);
    __syncthreads();
  }
  m = red[0];
  __syncthreads();
  float e[4], sum = 0.f;
#pragma unroll
  for (int t = 0; t < 4; ++t) { e[t] = __expf(v[t] - m); sum += e[t]; }
  red[tid] = sum;
  __syncthreads();
  for (int s = 128; s > 0; s >>= 1) {
    if (tid < s) red[tid] += red[tid + s];
    __syncthreads();
  }
  const float inv = 1.0f / red[0];
#pragma unroll
  for (int t = 0; t < 4; ++t) p[tid + 256 * t] = (bf16)(e[t] * inv);
}

// ---------------------------------------------------------------------------
// LayerNorm over channels + transpose back:  y[B,N,C] f32 -> out[B,C,N] f32.
// One block handles 32 tokens; padded LDS tile makes both sides coalesced.
// ---------------------------------------------------------------------------
__global__ __launch_bounds__(256)
void ln_kernel(const float* __restrict__ y, const float* __restrict__ gamma,
               const float* __restrict__ beta, float* __restrict__ out) {
  __shared__ float tile[32][257];
  __shared__ float ps[32][8], psq[32][8];
  __shared__ float mu[32], rstd[32];
  const int b   = blockIdx.x >> 5;          // 32 token-blocks per batch
  const int n0  = (blockIdx.x & 31) * 32;
  const int tid = threadIdx.x;
  const float* yb = y + ((size_t)b * NT + n0) * C;
#pragma unroll 4
  for (int r = 0; r < 32; ++r) tile[r][tid] = yb[(size_t)r * C + tid];
  __syncthreads();
  const int rr = tid >> 3, g = tid & 7;
  float s = 0.f, sq = 0.f;
  for (int c = g; c < C; c += 8) { float v = tile[rr][c]; s += v; sq += v * v; }
  ps[rr][g] = s; psq[rr][g] = sq;
  __syncthreads();
  if (g == 0) {
    float S = 0.f, Q = 0.f;
#pragma unroll
    for (int i = 0; i < 8; ++i) { S += ps[rr][i]; Q += psq[rr][i]; }
    float mean = S * (1.0f / C);
    float var  = Q * (1.0f / C) - mean * mean;
    mu[rr] = mean;
    rstd[rr] = rsqrtf(var + 1e-5f);
  }
  __syncthreads();
  const float gm = gamma[tid], bt = beta[tid];
#pragma unroll 4
  for (int r = 0; r < 32; ++r)
    tile[r][tid] = (tile[r][tid] - mu[r]) * rstd[r] * gm + bt;
  __syncthreads();
  float* ob = out + (size_t)b * C * NT + n0;
#pragma unroll
  for (int cb = 0; cb < 8; ++cb) {
    const int c  = cb * 32 + (tid >> 5);
    const int nl = tid & 31;
    ob[(size_t)c * NT + nl] = tile[nl][c];
  }
}

// ---------------------------------------------------------------------------
extern "C" void kernel_launch(void* const* d_in, const int* in_sizes, int n_in,
                              void* d_out, int out_size, void* d_ws, size_t ws_size,
                              hipStream_t stream) {
  const float* x     = (const float*)d_in[0];
  const float* wq    = (const float*)d_in[1];
  const float* bq    = (const float*)d_in[2];
  const float* wk    = (const float*)d_in[3];
  const float* bk    = (const float*)d_in[4];
  const float* wv    = (const float*)d_in[5];
  const float* bv    = (const float*)d_in[6];
  const float* w1    = (const float*)d_in[7];
  const float* b1    = (const float*)d_in[8];
  const float* w2    = (const float*)d_in[9];
  const float* b2    = (const float*)d_in[10];
  const float* gamma = (const float*)d_in[11];
  const float* beta  = (const float*)d_in[12];
  float* out = (float*)d_out;

  // Workspace carve-up (256B aligned), ~99 MB total.
  char* ws = (char*)d_ws;
  size_t off = 0;
  auto carve = [&](size_t bytes) {
    char* p = ws + off;
    off = (off + bytes + 255) & ~(size_t)255;
    return p;
  };
  bf16*  tbf  = (bf16*)carve((size_t)Bsz * NT * C * 2);       // x transposed, bf16
  bf16*  wt   = (bf16*)carve((size_t)5 * C * C * 2);          // packed weights^T
  bf16*  qbf  = (bf16*)carve((size_t)Bsz * NT * C * 2);
  bf16*  kbf  = (bf16*)carve((size_t)Bsz * NT * C * 2);
  bf16*  vT   = (bf16*)carve((size_t)Bsz * C * NT * 2);       // [B,C,N]
  bf16*  S    = (bf16*)carve((size_t)Bsz * NT * NT * 2);      // scores -> probs
  bf16*  att0 = (bf16*)carve((size_t)Bsz * NT * C * 2);
  bf16*  hbuf = (bf16*)carve((size_t)Bsz * NT * C * 2);
  float* ybuf = (float*)carve((size_t)Bsz * NT * C * 4);

  const long long sNC = (long long)NT * C;
  const long long sNN = (long long)NT * NT;
  const long long sCN = (long long)C * NT;

  // 1) transpose + bf16 convert input
  xpose_kernel<<<dim3(NT / 32, C / 32, Bsz), 256, 0, stream>>>(x, tbf);
  // 2) pack weights
  wpack_kernel<<<dim3(C * C / 256, 5), 256, 0, stream>>>(wq, wk, wv, w1, w2, wt);

  // 3) QKV projections (M=1024, N=256, K=256); block tile 64x128
  gemm_bt_kernel<0><<<dim3(C / 128, NT / 64, Bsz), 128, 0, stream>>>(
      tbf, wt + 0 * C * C, bq, qbf, C, C, C, C, sNC, 0, sNC);
  gemm_bt_kernel<0><<<dim3(C / 128, NT / 64, Bsz), 128, 0, stream>>>(
      tbf, wt + 1 * C * C, bk, kbf, C, C, C, C, sNC, 0, sNC);
  gemm_bt_kernel<1><<<dim3(C / 128, NT / 64, Bsz), 128, 0, stream>>>(
      tbf, wt + 2 * C * C, bv, vT, C, C, C, NT, sNC, 0, sCN);

  // 4) scores: S[j,i] = k[j,:] . q[i,:]   (softmax axis of reference == i)
  gemm_bt_kernel<2><<<dim3(NT / 128, NT / 64, Bsz), 128, 0, stream>>>(
      kbf, qbf, nullptr, S, C, C, C, NT, sNC, sNC, sNN);

  // 5) softmax over i (row-wise, in place, bf16)
  softmax_kernel<<<Bsz * NT, 256, 0, stream>>>(S);

  // 6) att0[j,c] = sum_i P[j,i] * v[i,c]   (B^T buffer is vT[C,N])
  gemm_bt_kernel<2><<<dim3(C / 128, NT / 64, Bsz), 128, 0, stream>>>(
      S, vT, nullptr, att0, NT, NT, NT, C, sNN, sCN, sNC);

  // 7) MLP: h = silu(att0 @ w1 + b1);  y = h @ w2 + b2 (f32)
  gemm_bt_kernel<3><<<dim3(C / 128, NT / 64, Bsz), 128, 0, stream>>>(
      att0, wt + 3 * C * C, b1, hbuf, C, C, C, C, sNC, 0, sNC);
  gemm_bt_kernel<4><<<dim3(C / 128, NT / 64, Bsz), 128, 0, stream>>>(
      hbuf, wt + 4 * C * C, b2, ybuf, C, C, C, C, sNC, 0, sNC);

  // 8) LayerNorm over channels + transpose back to [B,C,H,W]
  ln_kernel<<<Bsz * (NT / 32), 256, 0, stream>>>(ybuf, gamma, beta, out);
}
